// MaskGatedMoEFFN_83811991814297
// MI455X (gfx1250) — compile-verified, matrix operand order checked by reading, not run on hardware
//
#include <hip/hip_runtime.h>
#include <math.h>

// ---------------- types for CDNA5 WMMA ----------------
typedef __attribute__((ext_vector_type(16))) __bf16 bf16x16;
typedef __attribute__((ext_vector_type(8)))  float  f32x8;

__device__ __forceinline__ unsigned short f2bf(float f) {
    unsigned int u = __float_as_uint(f);
    unsigned int r = (u + 0x7FFFu + ((u >> 16) & 1u)) >> 16;
    return (unsigned short)r;
}

__device__ __forceinline__ bf16x16 lds_frag(const unsigned short* p) {
    union { uint4 q[2]; bf16x16 v; } u;
    u.q[0] = *(const uint4*)(p);
    u.q[1] = *(const uint4*)(p + 16);   // +32 bytes -> K+16..K+23 chunk
    return u.v;
}

// Async 16B global->LDS copy (per lane), tracked by ASYNCcnt.
// LDS byte address = low 32 bits of the generic pointer (flat aperture rule).
__device__ __forceinline__ void async_copy_b128(const unsigned short* lds_dst,
                                                const unsigned short* g_src) {
    unsigned       loff = (unsigned)(unsigned long long)lds_dst;
    unsigned long long ga = (unsigned long long)g_src;
    asm volatile("global_load_async_to_lds_b128 %0, %1, off"
                 :: "v"(loff), "v"(ga) : "memory");
}

__device__ __forceinline__ void wait_async0() {
#if __has_builtin(__builtin_amdgcn_s_wait_asynccnt)
    __builtin_amdgcn_s_wait_asynccnt(0);
#else
    asm volatile("s_wait_asynccnt 0x0" ::: "memory");
#endif
}

// ---------------- gating network: conv3d + bn + relu ----------------
// conv1: in mask [1024][1][16][32][32], out [1024][32][8][16][16]
__global__ void conv1_bn_relu(const float* __restrict__ in, const float* __restrict__ w,
                              const float* __restrict__ bias,
                              const float* __restrict__ bng, const float* __restrict__ bnb,
                              const float* __restrict__ bnm, const float* __restrict__ bnv,
                              float* __restrict__ out) {
    int blk = blockIdx.x;                 // 1024*32*8 blocks
    int b   = blk >> 8;
    int oc  = (blk >> 3) & 31;
    int p   = ((blk & 7) << 8) | threadIdx.x;   // 0..2047
    int z = p >> 8, y = (p >> 4) & 15, x = p & 15;
    const float* wp = w + oc * 27;
    const float* inb = in + (size_t)b * (16 * 32 * 32);
    float sum = bias[oc];
    #pragma unroll
    for (int kd = 0; kd < 3; kd++) {
        int zi = 2 * z - 1 + kd; if ((unsigned)zi >= 16u) continue;
        #pragma unroll
        for (int kh = 0; kh < 3; kh++) {
            int yi = 2 * y - 1 + kh; if ((unsigned)yi >= 32u) continue;
            #pragma unroll
            for (int kw = 0; kw < 3; kw++) {
                int xi = 2 * x - 1 + kw; if ((unsigned)xi >= 32u) continue;
                sum += inb[(zi * 32 + yi) * 32 + xi] * wp[(kd * 3 + kh) * 3 + kw];
            }
        }
    }
    float scale = bng[oc] * rsqrtf(bnv[oc] + 1e-5f);
    float h = (sum - bnm[oc]) * scale + bnb[oc];
    out[(size_t)(blk >> 3) * 2048 + p] = fmaxf(h, 0.0f);
}

// conv2: in [1024][32][8][16][16], out [1024][64][4][8][8]
__global__ void conv2_bn_relu(const float* __restrict__ in, const float* __restrict__ w,
                              const float* __restrict__ bias,
                              const float* __restrict__ bng, const float* __restrict__ bnb,
                              const float* __restrict__ bnm, const float* __restrict__ bnv,
                              float* __restrict__ out) {
    __shared__ float wsh[864];
    int b  = blockIdx.x >> 6;
    int oc = blockIdx.x & 63;
    for (int t = threadIdx.x; t < 864; t += 256) wsh[t] = w[oc * 864 + t];
    __syncthreads();
    int p = threadIdx.x;                    // 0..255
    int z = p >> 6, y = (p >> 3) & 7, x = p & 7;
    float sum = bias[oc];
    const float* inb = in + (size_t)b * (32 * 2048);
    for (int ic = 0; ic < 32; ic++) {
        const float* ini = inb + ic * 2048;
        const float* wi  = wsh + ic * 27;
        #pragma unroll
        for (int kd = 0; kd < 3; kd++) {
            int zi = 2 * z - 1 + kd; if ((unsigned)zi >= 8u) continue;
            #pragma unroll
            for (int kh = 0; kh < 3; kh++) {
                int yi = 2 * y - 1 + kh; if ((unsigned)yi >= 16u) continue;
                #pragma unroll
                for (int kw = 0; kw < 3; kw++) {
                    int xi = 2 * x - 1 + kw; if ((unsigned)xi >= 16u) continue;
                    sum += ini[(zi * 16 + yi) * 16 + xi] * wi[(kd * 3 + kh) * 3 + kw];
                }
            }
        }
    }
    float scale = bng[oc] * rsqrtf(bnv[oc] + 1e-5f);
    float h = (sum - bnm[oc]) * scale + bnb[oc];
    out[(size_t)blockIdx.x * 256 + p] = fmaxf(h, 0.0f);
}

// conv3: in [1024][64][4][8][8], out [1024][128][2][4][4]
__global__ void conv3_bn_relu(const float* __restrict__ in, const float* __restrict__ w,
                              const float* __restrict__ bias,
                              const float* __restrict__ bng, const float* __restrict__ bnb,
                              const float* __restrict__ bnm, const float* __restrict__ bnv,
                              float* __restrict__ out) {
    __shared__ float wsh[1728];
    int oc = blockIdx.y;
    for (int t = threadIdx.x; t < 1728; t += 256) wsh[t] = w[oc * 1728 + t];
    __syncthreads();
    int bsub = threadIdx.x >> 5;
    int b = (blockIdx.x << 3) + bsub;
    int p = threadIdx.x & 31;
    int z = p >> 4, y = (p >> 2) & 3, x = p & 3;
    float sum = bias[oc];
    const float* inb = in + (size_t)b * (64 * 256);
    for (int ic = 0; ic < 64; ic++) {
        const float* ini = inb + ic * 256;
        const float* wi  = wsh + ic * 27;
        #pragma unroll
        for (int kd = 0; kd < 3; kd++) {
            int zi = 2 * z - 1 + kd; if ((unsigned)zi >= 4u) continue;
            #pragma unroll
            for (int kh = 0; kh < 3; kh++) {
                int yi = 2 * y - 1 + kh; if ((unsigned)yi >= 8u) continue;
                #pragma unroll
                for (int kw = 0; kw < 3; kw++) {
                    int xi = 2 * x - 1 + kw; if ((unsigned)xi >= 8u) continue;
                    sum += ini[(zi * 8 + yi) * 8 + xi] * wi[(kd * 3 + kh) * 3 + kw];
                }
            }
        }
    }
    float scale = bng[oc] * rsqrtf(bnv[oc] + 1e-5f);
    float h = (sum - bnm[oc]) * scale + bnb[oc];
    out[((size_t)b * 128 + oc) * 32 + p] = fmaxf(h, 0.0f);
}

// pool [1024][128][32] -> mean -> fc(4x128) -> softmax -> g [1024][4]
__global__ void pool_fc_softmax(const float* __restrict__ in, const float* __restrict__ fcw,
                                const float* __restrict__ fcb, float* __restrict__ g) {
    __shared__ float pooled[128];
    __shared__ float logits[4];
    int b = blockIdx.x, c = threadIdx.x;
    const float* pi = in + ((size_t)b * 128 + c) * 32;
    float s = 0.f;
    #pragma unroll
    for (int i = 0; i < 32; i++) s += pi[i];
    pooled[c] = s * (1.0f / 32.0f);
    __syncthreads();
    if (c < 4) {
        float l = fcb[c];
        for (int i = 0; i < 128; i++) l += fcw[c * 128 + i] * pooled[i];
        logits[c] = l;
    }
    __syncthreads();
    if (c == 0) {
        float m = fmaxf(fmaxf(logits[0], logits[1]), fmaxf(logits[2], logits[3]));
        float e0 = expf(logits[0] - m), e1 = expf(logits[1] - m);
        float e2 = expf(logits[2] - m), e3 = expf(logits[3] - m);
        float inv = 1.0f / (e0 + e1 + e2 + e3);
        g[b * 4 + 0] = e0 * inv; g[b * 4 + 1] = e1 * inv;
        g[b * 4 + 2] = e2 * inv; g[b * 4 + 3] = e3 * inv;
    }
}

// ---------------- fp32 -> bf16 conversions ----------------
__global__ void cvt_f32_bf16(const float* __restrict__ src, unsigned short* __restrict__ dst, int n) {
    int i = blockIdx.x * 256 + threadIdx.x;
    if (i < n) dst[i] = f2bf(src[i]);
}

// src (per e) [R][C] fp32 -> dst[(col)*dstRowStride + row + e*dstEStep] bf16 (transpose)
__global__ void transpose_cvt_bf16(const float* __restrict__ src, unsigned short* __restrict__ dst,
                                   int R, int C, size_t dstRowStride, size_t dstEStep) {
    __shared__ float tile[32][33];
    size_t e = blockIdx.z;
    src += e * (size_t)R * C;
    unsigned short* d = dst + e * dstEStep;
    int c0 = blockIdx.x * 32;
    int r0 = blockIdx.y * 32;
    int tx = threadIdx.x, ty = threadIdx.y;
    for (int j = ty; j < 32; j += 8)
        tile[j][tx] = src[(size_t)(r0 + j) * C + (c0 + tx)];
    __syncthreads();
    for (int j = ty; j < 32; j += 8) {
        int c = c0 + j, r = r0 + tx;
        d[(size_t)c * dstRowStride + r] = f2bf(tile[tx][j]);
    }
}

// ---------------- GEMM1 + GEGLU + gate pre-scale ----------------
__device__ __forceinline__ void store_geglu(f32x8 v, f32x8 q, int row0, int col, int e,
                                            const float* __restrict__ b1,
                                            const float* __restrict__ g,
                                            unsigned short* __restrict__ hid) {
    float bv = b1[e * 10240 + col];
    float bg = b1[e * 10240 + 5120 + col];
    #pragma unroll
    for (int j = 0; j < 8; j++) {
        int row = row0 + j;
        float val = v[j] + bv;
        float gt  = q[j] + bg;
        float gel = 0.5f * gt * (1.0f + erff(gt * 0.70710678118654752f));
        float h   = val * gel * g[row * 4 + e];
        hid[(size_t)row * 20480 + e * 5120 + col] = f2bf(h);
    }
}

__global__ __launch_bounds__(256) void gemm1_geglu_kernel(
    const unsigned short* __restrict__ xbf,    // [1024][1280] bf16
    const unsigned short* __restrict__ w1t,    // [4][10240][1280] bf16 (k-major)
    const float* __restrict__ b1,              // [4][10240]
    const float* __restrict__ gating,          // [1024][4]
    unsigned short* __restrict__ hid)          // [1024][20480] bf16
{
    __shared__ __align__(16) unsigned short As[2][128 * 48];
    __shared__ __align__(16) unsigned short Bv[2][64 * 48];
    __shared__ __align__(16) unsigned short Bg[2][64 * 48];

    const int e     = blockIdx.z;
    const int iBase = blockIdx.x * 64;
    const int nBase = blockIdx.y * 128;
    const unsigned short* wv = w1t + (size_t)e * 10240 * 1280 + (size_t)iBase * 1280;
    const unsigned short* wg = wv + (size_t)5120 * 1280;
    const unsigned short* xa = xbf + (size_t)nBase * 1280;

    const int tid  = threadIdx.x;
    const int lane = tid & 31;
    const int wave = tid >> 5;
    const int mOff = (wave >> 1) * 32;
    const int nOff = (wave & 1) * 32;

    f32x8 zero = {0.f,0.f,0.f,0.f,0.f,0.f,0.f,0.f};
    f32x8 accv00 = zero, accv01 = zero, accv10 = zero, accv11 = zero;
    f32x8 accg00 = zero, accg01 = zero, accg10 = zero, accg11 = zero;

    const int ar0 = tid >> 2;              // 0..63
    const int akc = (tid & 3) * 8;         // 0,8,16,24

    // async stage of one K=32 tile into buffer `s`
    auto stage = [&](int s, int k0) {
        async_copy_b128(&As[s][ar0 * 48 + akc],        &xa[(size_t)ar0 * 1280 + k0 + akc]);
        async_copy_b128(&As[s][(ar0 + 64) * 48 + akc], &xa[(size_t)(ar0 + 64) * 1280 + k0 + akc]);
        async_copy_b128(&Bv[s][ar0 * 48 + akc],        &wv[(size_t)ar0 * 1280 + k0 + akc]);
        async_copy_b128(&Bg[s][ar0 * 48 + akc],        &wg[(size_t)ar0 * 1280 + k0 + akc]);
    };

    stage(0, 0);
    for (int k0 = 0; k0 < 1280; k0 += 32) {
        const int buf = (k0 >> 5) & 1;
        wait_async0();          // my writes into `buf` are complete
        __syncthreads();        // everyone's writes visible; prev reads of buf^1 done
        if (k0 + 32 < 1280) stage(buf ^ 1, k0 + 32);

        const int fr = lane & 15;
        const int fk = (lane >> 4) * 8;
        bf16x16 a0  = lds_frag(&As[buf][(mOff + fr) * 48 + fk]);
        bf16x16 a1  = lds_frag(&As[buf][(mOff + 16 + fr) * 48 + fk]);
        bf16x16 bv0 = lds_frag(&Bv[buf][(nOff + fr) * 48 + fk]);
        bf16x16 bv1 = lds_frag(&Bv[buf][(nOff + 16 + fr) * 48 + fk]);
        bf16x16 bg0 = lds_frag(&Bg[buf][(nOff + fr) * 48 + fk]);
        bf16x16 bg1 = lds_frag(&Bg[buf][(nOff + 16 + fr) * 48 + fk]);
        accv00 = __builtin_amdgcn_wmma_f32_16x16x32_bf16(false, a0, false, bv0, (short)0, accv00, false, false);
        accv01 = __builtin_amdgcn_wmma_f32_16x16x32_bf16(false, a0, false, bv1, (short)0, accv01, false, false);
        accv10 = __builtin_amdgcn_wmma_f32_16x16x32_bf16(false, a1, false, bv0, (short)0, accv10, false, false);
        accv11 = __builtin_amdgcn_wmma_f32_16x16x32_bf16(false, a1, false, bv1, (short)0, accv11, false, false);
        accg00 = __builtin_amdgcn_wmma_f32_16x16x32_bf16(false, a0, false, bg0, (short)0, accg00, false, false);
        accg01 = __builtin_amdgcn_wmma_f32_16x16x32_bf16(false, a0, false, bg1, (short)0, accg01, false, false);
        accg10 = __builtin_amdgcn_wmma_f32_16x16x32_bf16(false, a1, false, bg0, (short)0, accg10, false, false);
        accg11 = __builtin_amdgcn_wmma_f32_16x16x32_bf16(false, a1, false, bg1, (short)0, accg11, false, false);
    }

    const int colLoc  = lane & 15;
    const int rowHalf = (lane >> 4) * 8;
    const int row0 = nBase + mOff + rowHalf;
    const int col0 = iBase + nOff + colLoc;
    store_geglu(accv00, accg00, row0,      col0,      e, b1, gating, hid);
    store_geglu(accv01, accg01, row0,      col0 + 16, e, b1, gating, hid);
    store_geglu(accv10, accg10, row0 + 16, col0,      e, b1, gating, hid);
    store_geglu(accv11, accg11, row0 + 16, col0 + 16, e, b1, gating, hid);
}

// ---------------- GEMM2 (concatenated experts) + gated bias ----------------
__device__ __forceinline__ void store_out(f32x8 a, int row0, int d,
                                          const float* __restrict__ b2,
                                          const float* __restrict__ g,
                                          float* __restrict__ out) {
    #pragma unroll
    for (int j = 0; j < 8; j++) {
        int row = row0 + j;
        const float* gr = g + row * 4;
        float bias = gr[0] * b2[d] + gr[1] * b2[1280 + d] + gr[2] * b2[2560 + d] + gr[3] * b2[3840 + d];
        out[(size_t)row * 1280 + d] = a[j] + bias;
    }
}

__global__ __launch_bounds__(256) void gemm2_kernel(
    const unsigned short* __restrict__ hid,    // [1024][20480] bf16
    const unsigned short* __restrict__ w2t,    // [1280][20480] bf16
    const float* __restrict__ b2,              // [4][1280]
    const float* __restrict__ gating,          // [1024][4]
    float* __restrict__ out)                   // [1024][1280]
{
    __shared__ __align__(16) unsigned short As[2][128 * 48];
    __shared__ __align__(16) unsigned short Bs[2][64 * 48];

    const int dBase = blockIdx.x * 64;
    const int nBase = blockIdx.y * 128;
    const unsigned short* ha = hid + (size_t)nBase * 20480;
    const unsigned short* wb = w2t + (size_t)dBase * 20480;

    const int tid  = threadIdx.x;
    const int lane = tid & 31;
    const int wave = tid >> 5;
    const int mOff = (wave >> 1) * 32;
    const int nOff = (wave & 1) * 32;

    f32x8 zero = {0.f,0.f,0.f,0.f,0.f,0.f,0.f,0.f};
    f32x8 acc00 = zero, acc01 = zero, acc10 = zero, acc11 = zero;

    const int ar0 = tid >> 2;
    const int akc = (tid & 3) * 8;

    auto stage = [&](int s, int k0) {
        async_copy_b128(&As[s][ar0 * 48 + akc],        &ha[(size_t)ar0 * 20480 + k0 + akc]);
        async_copy_b128(&As[s][(ar0 + 64) * 48 + akc], &ha[(size_t)(ar0 + 64) * 20480 + k0 + akc]);
        async_copy_b128(&Bs[s][ar0 * 48 + akc],        &wb[(size_t)ar0 * 20480 + k0 + akc]);
    };

    stage(0, 0);
    for (int k0 = 0; k0 < 20480; k0 += 32) {
        const int buf = (k0 >> 5) & 1;
        wait_async0();
        __syncthreads();
        if (k0 + 32 < 20480) stage(buf ^ 1, k0 + 32);

        const int fr = lane & 15;
        const int fk = (lane >> 4) * 8;
        bf16x16 a0 = lds_frag(&As[buf][(mOff + fr) * 48 + fk]);
        bf16x16 a1 = lds_frag(&As[buf][(mOff + 16 + fr) * 48 + fk]);
        bf16x16 b0 = lds_frag(&Bs[buf][(nOff + fr) * 48 + fk]);
        bf16x16 b1 = lds_frag(&Bs[buf][(nOff + 16 + fr) * 48 + fk]);
        acc00 = __builtin_amdgcn_wmma_f32_16x16x32_bf16(false, a0, false, b0, (short)0, acc00, false, false);
        acc01 = __builtin_amdgcn_wmma_f32_16x16x32_bf16(false, a0, false, b1, (short)0, acc01, false, false);
        acc10 = __builtin_amdgcn_wmma_f32_16x16x32_bf16(false, a1, false, b0, (short)0, acc10, false, false);
        acc11 = __builtin_amdgcn_wmma_f32_16x16x32_bf16(false, a1, false, b1, (short)0, acc11, false, false);
    }

    const int colLoc  = lane & 15;
    const int rowHalf = (lane >> 4) * 8;
    const int row0 = nBase + mOff + rowHalf;
    const int col0 = dBase + nOff + colLoc;
    store_out(acc00, row0,      col0,      b2, gating, out);
    store_out(acc01, row0,      col0 + 16, b2, gating, out);
    store_out(acc10, row0 + 16, col0,      b2, gating, out);
    store_out(acc11, row0 + 16, col0 + 16, b2, gating, out);
}

// ---------------- launch ----------------
extern "C" void kernel_launch(void* const* d_in, const int* in_sizes, int n_in,
                              void* d_out, int out_size, void* d_ws, size_t ws_size,
                              hipStream_t stream) {
    const float* x     = (const float*)d_in[0];
    const float* mask  = (const float*)d_in[1];
    const float* c1w   = (const float*)d_in[2];
    const float* c1b   = (const float*)d_in[3];
    const float* bn1g  = (const float*)d_in[4];
    const float* bn1b  = (const float*)d_in[5];
    const float* bn1m  = (const float*)d_in[6];
    const float* bn1v  = (const float*)d_in[7];
    const float* c2w   = (const float*)d_in[8];
    const float* c2b   = (const float*)d_in[9];
    const float* bn2g  = (const float*)d_in[10];
    const float* bn2b  = (const float*)d_in[11];
    const float* bn2m  = (const float*)d_in[12];
    const float* bn2v  = (const float*)d_in[13];
    const float* c3w   = (const float*)d_in[14];
    const float* c3b   = (const float*)d_in[15];
    const float* bn3g  = (const float*)d_in[16];
    const float* bn3b  = (const float*)d_in[17];
    const float* bn3m  = (const float*)d_in[18];
    const float* bn3v  = (const float*)d_in[19];
    const float* fcw   = (const float*)d_in[20];
    const float* fcb   = (const float*)d_in[21];
    const float* w1    = (const float*)d_in[22];
    const float* b1    = (const float*)d_in[23];
    const float* w2    = (const float*)d_in[24];
    const float* b2    = (const float*)d_in[25];

    char* ws = (char*)d_ws;
    // Region A (268.4 MB): conv1 activations, later reused for bf16 weights + hidcat
    float*          out1 = (float*)(ws + 0);
    unsigned short* w1t  = (unsigned short*)(ws + 0);                 // 104857600 B
    unsigned short* w2t  = (unsigned short*)(ws + 104857600);         //  52428800 B
    unsigned short* xbf  = (unsigned short*)(ws + 104857600 + 52428800);          // 2621440 B
    unsigned short* hid  = (unsigned short*)(ws + 104857600 + 52428800 + 2621440);// 41943040 B
    // Region B
    float* out2 = (float*)(ws + 268435456);                 // 67108864 B
    float* out3 = (float*)(ws + 268435456 + 67108864);      // 16777216 B
    float* gbuf = (float*)(ws + 268435456 + 67108864 + 16777216); // 16384 B

    // --- gating network ---
    conv1_bn_relu<<<262144, 256, 0, stream>>>(mask, c1w, c1b, bn1g, bn1b, bn1m, bn1v, out1);
    conv2_bn_relu<<<65536, 256, 0, stream>>>(out1, c2w, c2b, bn2g, bn2b, bn2m, bn2v, out2);
    conv3_bn_relu<<<dim3(128, 128), 256, 0, stream>>>(out2, c3w, c3b, bn3g, bn3b, bn3m, bn3v, out3);
    pool_fc_softmax<<<1024, 128, 0, stream>>>(out3, fcw, fcb, gbuf);

    // --- bf16 conversion / transposition (safe: conv1 activations dead after conv2) ---
    cvt_f32_bf16<<<(1024 * 1280) / 256, 256, 0, stream>>>(x, xbf, 1024 * 1280);
    transpose_cvt_bf16<<<dim3(10240 / 32, 1280 / 32, 4), dim3(32, 8), 0, stream>>>(
        w1, w1t, 1280, 10240, 1280, (size_t)10240 * 1280);
    transpose_cvt_bf16<<<dim3(1280 / 32, 5120 / 32, 4), dim3(32, 8), 0, stream>>>(
        w2, w2t, 5120, 1280, 20480, 5120);

    // --- FFN: GEMM1+GEGLU (gate-prescaled), then single concatenated GEMM2 ---
    gemm1_geglu_kernel<<<dim3(5120 / 64, 1024 / 128, 4), 256, 0, stream>>>(
        xbf, w1t, b1, gbuf, hid);
    gemm2_kernel<<<dim3(1280 / 64, 1024 / 128), 256, 0, stream>>>(
        hid, w2t, b2, gbuf, (float*)d_out);

    (void)in_sizes; (void)n_in; (void)out_size; (void)ws_size;
}
